// Ponita_19602230739222
// MI455X (gfx1250) — matrix operand order repaired and checked by reading, not compiled, not run on hardware
//
#include <hip/hip_runtime.h>

#define NN 10000
#define EE 80000
#define ORI 12
#define CC 64
#define BASIS 64
#define IN_F 16
#define LL 3
#define BB 8
#define WF 4
#define PSP 14
#define EPSV 1e-6f
#define PI_F 3.14159265358979323846f

typedef __attribute__((ext_vector_type(16))) __bf16 v16bf;
typedef __attribute__((ext_vector_type(8)))  float  v8f;
typedef __attribute__((ext_vector_type(4)))  int    v4i;

#if __has_builtin(__builtin_amdgcn_global_load_async_to_lds_b128)
#define HAVE_ASYNC_LDS 1
typedef __attribute__((address_space(1))) v4i glob_v4i;
typedef __attribute__((address_space(3))) v4i lds_v4i;
#else
#define HAVE_ASYNC_LDS 0
#endif

// Branch-free tanh-approx gelu: tanh(u) = 1 - 2/(exp(2u)+1), exp via native v_exp_f32.
// Saturates correctly at +/-inf without branches.
__device__ __forceinline__ float gelu_f(float x) {
  float u = 0.7978845608028654f * (x + 0.044715f * x * x * x);
  float e = __builtin_amdgcn_exp2f(2.8853900817779268f * u);  // exp(2u)
  float r = __builtin_amdgcn_rcpf(e + 1.0f);
  float t = 1.0f - 2.0f * r;
  return 0.5f * x * (1.0f + t);
}

// Guaranteed-native fp32 scatter-add (no CAS fallback): GLOBAL_ATOMIC_ADD_F32, no-return form.
__device__ __forceinline__ void atomAddF(float* p, float v) {
  asm volatile("global_atomic_add_f32 %0, %1, off" :: "v"(p), "v"(v) : "memory");
}

// 16-byte global->LDS copy; async (ASYNCcnt-tracked) when the CDNA5 builtin is available.
__device__ __forceinline__ void copy16_to_lds(void* g, void* l) {
#if HAVE_ASYNC_LDS
  __builtin_amdgcn_global_load_async_to_lds_b128(
      (glob_v4i*)(void*)g, (lds_v4i*)(void*)l, 0, 0);
#else
  *(uint4*)l = *(const uint4*)g;
#endif
}

__device__ __forceinline__ void async_copy_wait() {
#if HAVE_ASYNC_LDS
  asm volatile("s_wait_asynccnt 0" ::: "memory");
#endif
}

__device__ __forceinline__ v8f wmma_bf16(v16bf a, v16bf b, v8f c) {
  return __builtin_amdgcn_wmma_f32_16x16x32_bf16(false, a, false, b, (short)0, c, false, false);
}

__device__ __forceinline__ v16bf zero_v16bf() {
  v16bf v;
#pragma unroll
  for (int i = 0; i < 16; ++i) v[i] = (__bf16)0.0f;
  return v;
}

// A fragment: 16x32 bf16 tile from row-major src (row stride `ld` halves).
__device__ __forceinline__ v16bf load_a_frag(const __bf16* base, int ld, int lane, int kbase) {
  int m  = lane & 15;
  int hi = (lane >> 4) & 1;
  const __bf16* rp = base + m * ld + kbase + hi * 8;
  union { uint4 u[2]; v16bf v; } f;
  f.u[0] = *(const uint4*)(rp);
  f.u[1] = *(const uint4*)(rp + 16);
  return f.v;
}

// B fragment: 32x16 bf16 tile (K x N), from f32 or bf16 row-major source.
__device__ __forceinline__ v16bf load_b_frag_f32(const float* w, int ld, int lane, int kbase, int nbase) {
  int n  = nbase + (lane & 15);
  int k0 = kbase + ((lane >> 4) << 3);
  v16bf f;
#pragma unroll
  for (int t = 0; t < 16; ++t) f[t] = (__bf16)w[(k0 + t + ((t >> 3) << 3)) * ld + n];
  return f;
}

__device__ __forceinline__ v16bf load_b_frag_bf(const __bf16* w, int ld, int lane, int kbase, int nbase) {
  int n  = nbase + (lane & 15);
  int k0 = kbase + ((lane >> 4) << 3);
  v16bf f;
#pragma unroll
  for (int t = 0; t < 16; ++t) f[t] = w[(k0 + t + ((t >> 3) << 3)) * ld + n];
  return f;
}

// ---------------- K0: ori_grid, fiber basis fk, bf16 l1_w copy, zero accumulators -------------
__global__ __launch_bounds__(256) void k0_setup(
    float* __restrict__ ori, float* __restrict__ fkb, float* __restrict__ fk,
    float* __restrict__ ro_acc, float* __restrict__ dout,
    const float* __restrict__ fb_w1, const float* __restrict__ fb_b1,
    const float* __restrict__ fb_w2, const float* __restrict__ fb_b2,
    const float* __restrict__ fib_kw,
    const float* __restrict__ l1_w, __bf16* __restrict__ l1wb) {
  int t = threadIdx.x;
  if (t < ORI) {
    const float GOLD = 3.23606797749979f;  // 1+sqrt(5)
    float th = fmodf(PI_F * (float)t * GOLD, 2.0f * PI_F);
    float ph = acosf(1.0f - 2.0f * ((float)t + 0.5f) / 12.0f);
    ori[t * 3 + 0] = sinf(ph) * cosf(th);
    ori[t * 3 + 1] = sinf(ph) * sinf(th);
    ori[t * 3 + 2] = cosf(ph);
  }
  __syncthreads();
  if (t < ORI * ORI) {
    int p = t / ORI, o = t % ORI;
    float d = ori[p * 3] * ori[o * 3] + ori[p * 3 + 1] * ori[o * 3 + 1] + ori[p * 3 + 2] * ori[o * 3 + 2];
    float pf0 = d, pf1 = d * d, pf2 = d * d * d;
    float hid[CC];
    for (int c = 0; c < CC; ++c)
      hid[c] = gelu_f(pf0 * fb_w1[0 * CC + c] + pf1 * fb_w1[1 * CC + c] + pf2 * fb_w1[2 * CC + c] + fb_b1[c]);
    for (int c = 0; c < CC; ++c) {
      float a = fb_b2[c];
      for (int k = 0; k < CC; ++k) a += hid[k] * fb_w2[k * CC + c];
      fkb[t * CC + c] = gelu_f(a);
    }
  }
  __syncthreads();
  for (int idx = t; idx < LL * ORI * ORI * CC; idx += blockDim.x) {
    int c  = idx & 63;
    int po = (idx >> 6) % (ORI * ORI);
    int i  = idx / (CC * ORI * ORI);
    float a = 0.f;
    for (int b = 0; b < BASIS; ++b) a += fkb[po * CC + b] * fib_kw[i * BASIS * CC + b * CC + c];
    fk[idx] = a;
  }
  for (int idx = t; idx < LL * CC * WF * CC; idx += blockDim.x) l1wb[idx] = (__bf16)l1_w[idx];
  for (int idx = t; idx < NN * ORI * 2; idx += blockDim.x) ro_acc[idx] = 0.f;
  for (int idx = t; idx < BB + BB * 3; idx += blockDim.x) dout[idx] = 0.f;
}

// ---------------- K1: h = broadcast(x @ emb_w) over orientations ----------------
__global__ __launch_bounds__(256) void k1_embed(const float* __restrict__ x,
                                                const float* __restrict__ emb_w,
                                                float* __restrict__ h) {
  int idx = blockIdx.x * blockDim.x + threadIdx.x;  // node*64 + c
  if (idx >= NN * CC) return;
  int node = idx >> 6, c = idx & 63;
  float a = 0.f;
#pragma unroll
  for (int k = 0; k < IN_F; ++k) a += x[node * IN_F + k] * emb_w[k * CC + c];
  for (int o = 0; o < ORI; ++o) h[node * ORI * CC + o * CC + c] = a;
}

// ---------------- K2: edge invariants -> poly feats -> MLP(14->64->64) -> kb (bf16) -----------
__global__ __launch_bounds__(256) void k2_kbasis(
    const float* __restrict__ pos, const int* __restrict__ e0, const int* __restrict__ e1,
    const float* __restrict__ oriw,
    const float* __restrict__ sb_w1, const float* __restrict__ sb_b1,
    const float* __restrict__ sb_w2, const float* __restrict__ sb_b2,
    __bf16* __restrict__ kb) {
  __shared__ alignas(16) float w1s[PSP * CC];   // 3584 B = 224 x 16B chunks
  __shared__ float b1s[CC];
  __shared__ float oris[ORI * 3];
  __shared__ alignas(16) __bf16 stage[8][32][CC];
  int tid = threadIdx.x, lane = tid & 31, wv = tid >> 5;
  if (tid < PSP * CC / 4) copy16_to_lds((void*)(sb_w1 + tid * 4), &w1s[tid * 4]);
  if (tid < CC) b1s[tid] = sb_b1[tid];
  if (tid < ORI * 3) oris[tid] = oriw[tid];
  // loop-invariant B fragments of sb_w2 (64x64) held in registers
  v16bf Bf[4][2];
#pragma unroll
  for (int nt = 0; nt < 4; ++nt)
#pragma unroll
    for (int ks = 0; ks < 2; ++ks) Bf[nt][ks] = load_b_frag_f32(sb_w2, CC, lane, ks * 32, nt * 16);
  async_copy_wait();
  __syncthreads();

  int rowbase32 = (blockIdx.x * 8 + wv) * 32;
  int row = rowbase32 + lane;
  int e = row / ORI, o = row - e * ORI;
  int s = e0[e], d = e1[e];
  float rx = pos[s * 3 + 0] - pos[d * 3 + 0];
  float ry = pos[s * 3 + 1] - pos[d * 3 + 1];
  float rz = pos[s * 3 + 2] - pos[d * 3 + 2];
  float ox = oris[o * 3], oy = oris[o * 3 + 1], oz = oris[o * 3 + 2];
  float i1 = rx * ox + ry * oy + rz * oz;
  float px = rx - i1 * ox, py = ry - i1 * oy, pz = rz - i1 * oz;
  float i2 = sqrtf(px * px + py * py + pz * pz);
  float pf[PSP];
  pf[0] = i1; pf[1] = i2;
  pf[2] = i1 * i1; pf[3] = i1 * i2; pf[4] = i2 * i1; pf[5] = i2 * i2;
#pragma unroll
  for (int j = 0; j < 4; ++j) { pf[6 + 2 * j] = pf[2 + j] * i1; pf[7 + 2 * j] = pf[2 + j] * i2; }
#pragma unroll 4
  for (int c = 0; c < CC; ++c) {
    float a = b1s[c];
#pragma unroll
    for (int k = 0; k < PSP; ++k) a += pf[k] * w1s[k * CC + c];
    stage[wv][lane][c] = (__bf16)gelu_f(a);
  }
  __syncthreads();
  int nlane = lane & 15, mb = (lane >> 4) * 8;
#pragma unroll
  for (int half = 0; half < 2; ++half) {
    const __bf16* abase = &stage[wv][half * 16][0];
    v16bf a0 = load_a_frag(abase, CC, lane, 0);
    v16bf a1 = load_a_frag(abase, CC, lane, 32);
#pragma unroll
    for (int nt = 0; nt < 4; ++nt) {
      v8f acc = {};
      acc = wmma_bf16(a0, Bf[nt][0], acc);
      acc = wmma_bf16(a1, Bf[nt][1], acc);
      int ch = nt * 16 + nlane;
      float b2 = sb_b2[ch];
#pragma unroll
      for (int r = 0; r < 8; ++r) {
        int grow = rowbase32 + half * 16 + mb + r;
        kb[(size_t)grow * CC + ch] = (__bf16)gelu_f(acc[r] + b2);
      }
    }
  }
}

// ---------------- zero helper ----------------
__global__ __launch_bounds__(256) void k_zero4(float4* __restrict__ p, int n4) {
  int i = blockIdx.x * blockDim.x + threadIdx.x;
  if (i < n4) p[i] = make_float4(0.f, 0.f, 0.f, 0.f);
}

// ---------------- K4: msg = h[e0] * (kb @ conv_kw[i]); atomic scatter-add into x1[e1] ----------
__global__ __launch_bounds__(256) void k4_msg(
    const __bf16* __restrict__ kb, const float* __restrict__ conv_kw_i,
    const int* __restrict__ e0, const int* __restrict__ e1,
    const float* __restrict__ h, float* __restrict__ x1) {
  int tid = threadIdx.x, lane = tid & 31, wv = tid >> 5;
  v16bf Bf[4][2];
#pragma unroll
  for (int nt = 0; nt < 4; ++nt)
#pragma unroll
    for (int ks = 0; ks < 2; ++ks) Bf[nt][ks] = load_b_frag_f32(conv_kw_i, CC, lane, ks * 32, nt * 16);
  int tile = blockIdx.x * 8 + wv;
  int rowbase = tile * 16;
  const __bf16* abase = kb + (size_t)rowbase * CC;
  v16bf a0 = load_a_frag(abase, CC, lane, 0);
  v16bf a1 = load_a_frag(abase, CC, lane, 32);
  int nlane = lane & 15, mb = (lane >> 4) * 8;
#pragma unroll
  for (int nt = 0; nt < 4; ++nt) {
    v8f acc = {};
    acc = wmma_bf16(a0, Bf[nt][0], acc);
    acc = wmma_bf16(a1, Bf[nt][1], acc);
    int ch = nt * 16 + nlane;
#pragma unroll
    for (int r = 0; r < 8; ++r) {
      int grow = rowbase + mb + r;
      int e = grow / ORI, o = grow - e * ORI;
      int s = e0[e], d = e1[e];
      float hv = h[s * ORI * CC + o * CC + ch];
      atomAddF(&x1[d * ORI * CC + o * CC + ch], acc[r] * hv);
    }
  }
}

// ---------------- K5: orientation mixing + /ORI + bias + LayerNorm -> xn bf16 -----------------
__global__ __launch_bounds__(384) void k5_mix(
    const float* __restrict__ x1, const float* __restrict__ fk_i,
    const float* __restrict__ conv_b_i, const float* __restrict__ ln_s_i,
    const float* __restrict__ ln_b_i, __bf16* __restrict__ xn) {
  __shared__ float x1s[ORI * CC];
  int node = blockIdx.x;
  int tid = threadIdx.x, lane = tid & 31, p = tid >> 5;  // 12 waves = 12 output orientations
  for (int i = tid; i < ORI * CC; i += blockDim.x) x1s[i] = x1[node * ORI * CC + i];
  __syncthreads();
  const float* fkp = fk_i + p * ORI * CC;
  int c0 = lane * 2, c1 = c0 + 1;
  float a0 = 0.f, a1 = 0.f;
#pragma unroll
  for (int o = 0; o < ORI; ++o) {
    a0 += x1s[o * CC + c0] * fkp[o * CC + c0];
    a1 += x1s[o * CC + c1] * fkp[o * CC + c1];
  }
  a0 = a0 * (1.0f / ORI) + conv_b_i[c0];
  a1 = a1 * (1.0f / ORI) + conv_b_i[c1];
  float s1 = a0 + a1, s2 = a0 * a0 + a1 * a1;
#pragma unroll
  for (int m = 16; m >= 1; m >>= 1) { s1 += __shfl_xor(s1, m, 32); s2 += __shfl_xor(s2, m, 32); }
  float mu  = s1 * (1.0f / CC);
  float var = s2 * (1.0f / CC) - mu * mu;
  float rs  = __builtin_amdgcn_rsqf(var + EPSV);
  float n0 = (a0 - mu) * rs * ln_s_i[c0] + ln_b_i[c0];
  float n1 = (a1 - mu) * rs * ln_s_i[c1] + ln_b_i[c1];
  int base = node * ORI * CC + p * CC;
  xn[base + c0] = (__bf16)n0;
  xn[base + c1] = (__bf16)n1;
}

// ---------------- K6: h += l2_b + (gelu(xn@l1_w + l1_b)) @ l2_w  (chained WMMA via LDS) -------
__global__ __launch_bounds__(256) void k6_mlp(
    const __bf16* __restrict__ xn, const __bf16* __restrict__ l1w_bf_i,
    const float* __restrict__ l1_b_i, const float* __restrict__ l2_w_i,
    const float* __restrict__ l2_b_i, float* __restrict__ h) {
  __shared__ alignas(16) __bf16 w1s[CC * WF * CC];      // 64x256 bf16 = 32KB
  __shared__ alignas(16) __bf16 hstage[8][16][CC];      // per-wave 16x64 stage = 16KB
  int tid = threadIdx.x, lane = tid & 31, wv = tid >> 5;
  for (int c = tid; c < CC * WF * CC / 8; c += blockDim.x)   // 2048 x 16B chunks
    copy16_to_lds((void*)(l1w_bf_i + c * 8), &w1s[c * 8]);
  async_copy_wait();
  __syncthreads();
  int tile = blockIdx.x * 8 + wv;
  bool valid = tile < (NN * ORI / 16);
  int rowbase = tile * 16;
  int nlane = lane & 15, mb = (lane >> 4) * 8;
  v16bf a0 = zero_v16bf(), a1 = zero_v16bf();
  if (valid) {
    const __bf16* abase = xn + (size_t)rowbase * CC;
    a0 = load_a_frag(abase, CC, lane, 0);
    a1 = load_a_frag(abase, CC, lane, 32);
  }
  v8f outacc[4] = {};
  for (int j = 0; j < WF; ++j) {
    if (valid) {
#pragma unroll
      for (int nt = 0; nt < 4; ++nt) {
        int col0 = j * CC + nt * 16;
        v8f acc = {};
        acc = wmma_bf16(a0, load_b_frag_bf(w1s, WF * CC, lane, 0, col0), acc);
        acc = wmma_bf16(a1, load_b_frag_bf(w1s, WF * CC, lane, 32, col0), acc);
        float bb = l1_b_i[col0 + nlane];
#pragma unroll
        for (int r = 0; r < 8; ++r) hstage[wv][mb + r][nt * 16 + nlane] = (__bf16)gelu_f(acc[r] + bb);
      }
    }
    __syncthreads();
    if (valid) {
      const __bf16* hb = &hstage[wv][0][0];
      v16bf h0 = load_a_frag(hb, CC, lane, 0);
      v16bf h1 = load_a_frag(hb, CC, lane, 32);
      const float* w2 = l2_w_i + j * CC * CC;  // rows j*64..j*64+63 of l2_w
#pragma unroll
      for (int nt = 0; nt < 4; ++nt) {
        outacc[nt] = wmma_bf16(h0, load_b_frag_f32(w2, CC, lane, 0, nt * 16), outacc[nt]);
        outacc[nt] = wmma_bf16(h1, load_b_frag_f32(w2, CC, lane, 32, nt * 16), outacc[nt]);
      }
    }
    __syncthreads();
  }
  if (valid) {
#pragma unroll
    for (int nt = 0; nt < 4; ++nt) {
      int ch = nt * 16 + nlane;
      float b2 = l2_b_i[ch];
#pragma unroll
      for (int r = 0; r < 8; ++r) {
        size_t idx = (size_t)(rowbase + mb + r) * CC + ch;
        h[idx] = h[idx] + outacc[nt][r] + b2;
      }
    }
  }
}

// ---------------- K7: ro_acc[row][0:2] += h[row] @ ro_w[i] + ro_b[i] ----------------
__global__ __launch_bounds__(256) void k7_readout(
    const float* __restrict__ h, const float* __restrict__ ro_w_i,
    const float* __restrict__ ro_b_i, float* __restrict__ ro_acc) {
  int tid = threadIdx.x, lane = tid & 31, wv = tid >> 5;
  int row = blockIdx.x * 8 + wv;
  if (row >= NN * ORI) return;
  int c0 = lane * 2;
  float h0 = h[(size_t)row * CC + c0], h1 = h[(size_t)row * CC + c0 + 1];
  float s = h0 * ro_w_i[c0 * 2 + 0] + h1 * ro_w_i[c0 * 2 + 2];
  float v = h0 * ro_w_i[c0 * 2 + 1] + h1 * ro_w_i[c0 * 2 + 3];
#pragma unroll
  for (int m = 16; m >= 1; m >>= 1) { s += __shfl_xor(s, m, 32); v += __shfl_xor(v, m, 32); }
  if (lane == 0) {
    ro_acc[row * 2 + 0] += s + ro_b_i[0];
    ro_acc[row * 2 + 1] += v + ro_b_i[1];
  }
}

// ---------------- K8: orientation mean / vector projection + batch segment-sum ----------------
__global__ __launch_bounds__(256) void k8_final(
    const float* __restrict__ ro_acc, const int* __restrict__ batch,
    const float* __restrict__ ori, float* __restrict__ dout) {
  int n = blockIdx.x * blockDim.x + threadIdx.x;
  if (n >= NN) return;
  float sa = 0.f, v0 = 0.f, v1 = 0.f, v2 = 0.f;
  for (int o = 0; o < ORI; ++o) {
    float s = ro_acc[(n * ORI + o) * 2 + 0];
    float v = ro_acc[(n * ORI + o) * 2 + 1];
    sa += s;
    v0 += v * ori[o * 3 + 0];
    v1 += v * ori[o * 3 + 1];
    v2 += v * ori[o * 3 + 2];
  }
  float inv = 1.0f / (float)(LL * ORI);
  int b = batch[n];
  atomAddF(&dout[b], sa * inv);
  atomAddF(&dout[BB + b * 3 + 0], v0 * inv);
  atomAddF(&dout[BB + b * 3 + 1], v1 * inv);
  atomAddF(&dout[BB + b * 3 + 2], v2 * inv);
}

extern "C" void kernel_launch(void* const* d_in, const int* in_sizes, int n_in,
                              void* d_out, int out_size, void* d_ws, size_t ws_size,
                              hipStream_t stream) {
  (void)in_sizes; (void)n_in; (void)out_size; (void)ws_size;
  const float* pos    = (const float*)d_in[0];
  const float* x      = (const float*)d_in[1];
  const int*   eidx   = (const int*)d_in[2];
  const int*   batch  = (const int*)d_in[3];
  const float* sb_w1  = (const float*)d_in[4];
  const float* sb_b1  = (const float*)d_in[5];
  const float* sb_w2  = (const float*)d_in[6];
  const float* sb_b2  = (const float*)d_in[7];
  const float* fb_w1  = (const float*)d_in[8];
  const float* fb_b1  = (const float*)d_in[9];
  const float* fb_w2  = (const float*)d_in[10];
  const float* fb_b2  = (const float*)d_in[11];
  const float* emb_w  = (const float*)d_in[12];
  const float* conv_kw = (const float*)d_in[13];
  const float* fib_kw  = (const float*)d_in[14];
  const float* conv_b  = (const float*)d_in[15];
  const float* ln_s    = (const float*)d_in[16];
  const float* ln_b    = (const float*)d_in[17];
  const float* l1_w    = (const float*)d_in[18];
  const float* l1_b    = (const float*)d_in[19];
  const float* l2_w    = (const float*)d_in[20];
  const float* l2_b    = (const float*)d_in[21];
  const float* ro_w    = (const float*)d_in[22];
  const float* ro_b    = (const float*)d_in[23];
  const int* e0 = eidx;
  const int* e1 = eidx + EE;
  float* out = (float*)d_out;

  char* w = (char*)d_ws;
  size_t off = 0;
  auto take = [&](size_t bytes) -> char* {
    char* p = w + off;
    off += (bytes + 255) & ~(size_t)255;
    return p;
  };
  float*  ori    = (float*)take(ORI * 3 * sizeof(float));
  float*  fkb    = (float*)take((size_t)ORI * ORI * CC * sizeof(float));
  float*  fk     = (float*)take((size_t)LL * ORI * ORI * CC * sizeof(float));
  float*  ro_acc = (float*)take((size_t)NN * ORI * 2 * sizeof(float));
  float*  h      = (float*)take((size_t)NN * ORI * CC * sizeof(float));
  float*  x1     = (float*)take((size_t)NN * ORI * CC * sizeof(float));
  __bf16* xn     = (__bf16*)take((size_t)NN * ORI * CC * sizeof(__bf16));
  __bf16* l1wb   = (__bf16*)take((size_t)LL * CC * WF * CC * sizeof(__bf16));
  __bf16* kb     = (__bf16*)take((size_t)EE * ORI * CC * sizeof(__bf16));

  k0_setup<<<1, 256, 0, stream>>>(ori, fkb, fk, ro_acc, out, fb_w1, fb_b1, fb_w2, fb_b2,
                                  fib_kw, l1_w, l1wb);
  k1_embed<<<(NN * CC + 255) / 256, 256, 0, stream>>>(x, emb_w, h);
  k2_kbasis<<<EE * ORI / 256, 256, 0, stream>>>(pos, e0, e1, ori, sb_w1, sb_b1, sb_w2, sb_b2, kb);

  for (int i = 0; i < LL; ++i) {
    k_zero4<<<NN * ORI * CC / 4 / 256, 256, 0, stream>>>((float4*)x1, NN * ORI * CC / 4);
    k4_msg<<<EE * ORI / 16 / 8, 256, 0, stream>>>(kb, conv_kw + (size_t)i * BASIS * CC, e0, e1, h, x1);
    k5_mix<<<NN, 384, 0, stream>>>(x1, fk + (size_t)i * ORI * ORI * CC, conv_b + i * CC,
                                   ln_s + i * CC, ln_b + i * CC, xn);
    k6_mlp<<<(NN * ORI / 16 + 7) / 8, 256, 0, stream>>>(xn, l1wb + (size_t)i * CC * WF * CC,
                                                        l1_b + i * WF * CC,
                                                        l2_w + (size_t)i * WF * CC * CC,
                                                        l2_b + i * CC, h);
    k7_readout<<<NN * ORI / 8, 256, 0, stream>>>(h, ro_w + i * CC * 2, ro_b + i * 2, ro_acc);
  }
  k8_final<<<(NN + 255) / 256, 256, 0, stream>>>(ro_acc, batch, ori, out);
}
